// BigramLanguageMode_7095285973153
// MI455X (gfx1250) — matrix operand order repaired and checked by reading, not compiled, not run on hardware
//
#include <hip/hip_runtime.h>

typedef _Float16 v16h __attribute__((ext_vector_type(16)));
typedef _Float16 v8h  __attribute__((ext_vector_type(8)));
typedef _Float16 h2   __attribute__((ext_vector_type(2)));
typedef float    v8f  __attribute__((ext_vector_type(8)));

// B=131072 seqs, T=8 -> 1,048,576 rows -> 65536 16-row tiles.
// grid 1024 blocks * 8 waves * 8 tiles/wave = 65536 tiles.
#define WPB   8     // waves per block
#define TPW   8     // tiles per wave
#define NBLK  1024
#define QROWS 528   // 520 (65 tok x 8 pos) qkv rows, padded to 33 full tiles

#define QSCALE      16.0f                       // q stored x16 (f16 denorm guard)
#define SCORE_SCALE (0.17677669529663687f / 16.0f)  // C^-0.5 / 16, C=32

union V8U { v8h v; h2 h[4]; };

// Compiler-level wave fence: CDNA5 LDS ops are in-order within a wave, so an
// intra-wave producer->consumer handoff only needs to stop compiler reordering.
__device__ __forceinline__ void wave_lds_sync() {
  __builtin_amdgcn_fence(__ATOMIC_ACQ_REL, "wavefront");
  __builtin_amdgcn_wave_barrier();
  __builtin_amdgcn_fence(__ATOMIC_ACQ_REL, "wavefront");
}

// 16x32 f16 A-fragment from row-major LDS (32 halves/row), contiguous-K layout:
// lanes 0-15:  row M=lane,    K in {0..15}
// lanes 16-31: row M=lane-16, K in {16..31}   -> two ds_load_b128
__device__ __forceinline__ v16h lds_load_afrag(const _Float16* p, int lane) {
  const int row = lane & 15;
  const int kh  = (lane >> 4) & 1;
  const v8h* c0 = reinterpret_cast<const v8h*>(p + row * 32 + kh * 16);
  const v8h* c1 = reinterpret_cast<const v8h*>(p + row * 32 + 8 + kh * 16);
  v8h lo = *c0, hi = *c1;
  v16h r;
#pragma unroll
  for (int i = 0; i < 8; ++i) { r[i] = lo[i]; r[i + 8] = hi[i]; }
  return r;
}

__global__ void __launch_bounds__(256, 1)
fused_bigram_attn_kernel(const int*   __restrict__ x,        // [B,8]
                         const float* __restrict__ tok_emb,  // [65,32]
                         const float* __restrict__ pos_emb,  // [8,32]
                         const float* __restrict__ Wq,       // [4,32,8]
                         const float* __restrict__ Wk,
                         const float* __restrict__ Wv,
                         const float* __restrict__ W1,       // [32,32]
                         const float* __restrict__ b1,       // [32]
                         const float* __restrict__ Wlm,      // [32,65]
                         const float* __restrict__ blm,      // [65]
                         float*       __restrict__ out)      // [B,8,65]
{
  // Shared QKV table: row (tok*8 + t) -> [16*q(32) | k(32) | v(32)] in f16.
  __shared__ __align__(16) _Float16 qkvTab[QROWS * 96];      // 101,376 B
  __shared__ __align__(16) _Float16 sB[WPB][16 * 32];        // attn out (f16)
  __shared__ __align__(16) _Float16 sA[WPB][16 * 32];        // relu out (f16)

  const int lane  = threadIdx.x & 31;
  // readfirstlane -> compiler-uniform wave id: token loads become s_load,
  // tile addressing goes scalar.
  const int wv    = __builtin_amdgcn_readfirstlane((int)(threadIdx.x >> 5));
  const int ncol  = lane & 15;               // B/C/D column within tile
  const int kb    = (lane & 16) ? 16 : 0;    // B-frag K-half base
  const int abase = (lane & 16) ? 8 : 0;     // half the A-frag K-half base (16)
  const int mofs  = (lane & 16) ? 8 : 0;     // C/D row offset

  // ============ One-time: build shared QKV table with WMMA ============
  {
    v16h fq[6];  // fused [16*Wq|Wk|Wv] as 32x96 B matrix fragments
#pragma unroll
    for (int j = 0; j < 6; ++j) {
      const int n  = j * 16 + ncol;          // 0..95
      const float* W = (n < 32) ? Wq : ((n < 64) ? Wk : Wv);
      const float sc = (n < 32) ? QSCALE : 1.0f;
      const int nn = n & 31, hh = nn >> 3, dd = nn & 7;
#pragma unroll
      for (int kk = 0; kk < 16; ++kk)
        fq[j][kk] = (_Float16)(W[(hh * 32 + (kb + kk)) * 8 + dd] * sc);
    }
    // 33 tiles of 16 rows over (tok,pos) pairs; waves split the tiles.
    for (int tb = wv; tb < 33; tb += WPB) {
      const int r  = tb * 16 + (lane & 15);      // 0..527
      const int tk = (r >> 3) > 64 ? 64 : (r >> 3);
      const int t  = r & 7;
      v16h a;                                    // h row, A-frag layout
#pragma unroll
      for (int i = 0; i < 8; ++i) {
        const int c0 = 2 * abase + i, c1 = 2 * abase + 8 + i;
        a[i]     = (_Float16)(tok_emb[tk * 32 + c0] + pos_emb[t * 32 + c0]);
        a[i + 8] = (_Float16)(tok_emb[tk * 32 + c1] + pos_emb[t * 32 + c1]);
      }
#pragma unroll
      for (int j = 0; j < 6; ++j) {
        v8f c = {};
        c = __builtin_amdgcn_wmma_f32_16x16x32_f16(false, a, false, fq[j],
                                                   (short)0, c, false, false);
#pragma unroll
        for (int vg = 0; vg < 8; ++vg)
          qkvTab[(tb * 16 + vg + mofs) * 96 + j * 16 + ncol] = (_Float16)c[vg];
      }
    }
  }

  // ============ Persistent per-wave weight fragments ============
  v16h f1[2];  // W1 32x32
#pragma unroll
  for (int j = 0; j < 2; ++j)
#pragma unroll
    for (int kk = 0; kk < 16; ++kk)
      f1[j][kk] = (_Float16)W1[(kb + kk) * 32 + (j * 16 + ncol)];
  v16h flm[5]; // Wlm 32x65 padded to 32x80
#pragma unroll
  for (int j = 0; j < 5; ++j) {
    const int n = j * 16 + ncol;
#pragma unroll
    for (int kk = 0; kk < 16; ++kk)
      flm[j][kk] = (n < 65) ? (_Float16)Wlm[(kb + kk) * 65 + n] : (_Float16)0.0f;
  }
  const float b1v[2] = { b1[ncol], b1[16 + ncol] };
  float blmv[5];
#pragma unroll
  for (int j = 0; j < 5; ++j) {
    const int n = j * 16 + ncol;
    blmv[j] = (n < 65) ? blm[n] : 0.0f;
  }

  __syncthreads();  // qkvTab visible block-wide; only block barrier needed.

  const int hd = lane >> 3;   // head 0..3
  const int tt = lane & 7;    // position 0..7
  const long tile0 = (long)(blockIdx.x * WPB + wv) * TPW;

  for (int it = 0; it < TPW; ++it) {
    const long rowbase = (tile0 + it) * 16;  // uniform flat row into [B*T]

    // ---- Attention: table-driven, packed-f16 math, unrolled softmax ----
#pragma unroll
    for (int s = 0; s < 2; ++s) {
      const int rb = s * 8;
      int kvbase[8];                         // uniform -> scalar loads/ALU
#pragma unroll
      for (int u = 0; u < 8; ++u)
        kvbase[u] = (x[rowbase + rb + u] * 8 + u) * 96;

      const int tokq = x[rowbase + rb + tt];         // per-lane (coalesced)
      V8U uq;
      uq.v = *(const v8h*)(qkvTab + (tokq * 8 + tt) * 96 + hd * 8);

      float sc[8];
      float mx = -1e30f;
#pragma unroll
      for (int u = 0; u < 8; ++u) {
        V8U uk;
        uk.v = *(const v8h*)(qkvTab + kvbase[u] + 32 + hd * 8);
        h2 acc = { (_Float16)0.0f, (_Float16)0.0f };
#pragma unroll
        for (int i = 0; i < 4; ++i) acc += uq.h[i] * uk.h[i];  // v_pk_fma_f16
        const float sf = ((float)acc[0] + (float)acc[1]) * SCORE_SCALE;
        sc[u] = (u <= tt) ? sf : -1e30f;    // causal mask, no divergence
        mx = fmaxf(mx, sc[u]);
      }
      float sum = 0.f;
#pragma unroll
      for (int u = 0; u < 8; ++u) { sc[u] = __expf(sc[u] - mx); sum += sc[u]; }
      const float inv = 1.f / sum;
      _Float16 p16[8];
#pragma unroll
      for (int u = 0; u < 8; ++u) p16[u] = (_Float16)(sc[u] * inv);

      h2 o4[4];
#pragma unroll
      for (int i = 0; i < 4; ++i) o4[i] = h2{ (_Float16)0.0f, (_Float16)0.0f };
#pragma unroll
      for (int u = 0; u < 8; ++u) {
        V8U uv;
        uv.v = *(const v8h*)(qkvTab + kvbase[u] + 64 + hd * 8);
        const h2 pp = { p16[u], p16[u] };
#pragma unroll
        for (int i = 0; i < 4; ++i) o4[i] += pp * uv.h[i];     // v_pk_fma_f16
      }
      V8U uo;
#pragma unroll
      for (int i = 0; i < 4; ++i) uo.h[i] = o4[i];
      *(v8h*)(sB[wv] + (rb + tt) * 32 + hd * 8) = uo.v;        // one b128
    }
    wave_lds_sync();

    // ---- relu(out @ W1 + b1): 2 WMMAs, bias folded into C ----
    {
      const v16h af = lds_load_afrag(sB[wv], lane);
#pragma unroll
      for (int j = 0; j < 2; ++j) {
        v8f c;
#pragma unroll
        for (int vg = 0; vg < 8; ++vg) c[vg] = b1v[j];
        c = __builtin_amdgcn_wmma_f32_16x16x32_f16(false, af, false, f1[j],
                                                   (short)0, c, false, false);
#pragma unroll
        for (int vg = 0; vg < 8; ++vg)
          sA[wv][(vg + mofs) * 32 + j * 16 + ncol] =
              (_Float16)fmaxf(c[vg], 0.f);
      }
    }
    wave_lds_sync();

    // ---- lm head (32x65): 5 WMMAs, bias in C, coalesced f32 stores ----
    {
      const v16h ar = lds_load_afrag(sA[wv], lane);
#pragma unroll
      for (int j = 0; j < 5; ++j) {
        v8f c;
#pragma unroll
        for (int vg = 0; vg < 8; ++vg) c[vg] = blmv[j];
        c = __builtin_amdgcn_wmma_f32_16x16x32_f16(false, ar, false, flm[j],
                                                   (short)0, c, false, false);
        const int col = j * 16 + ncol;
        if (col < 65) {
#pragma unroll
          for (int vg = 0; vg < 8; ++vg)
            out[(rowbase + vg + mofs) * 65 + col] = c[vg];
        }
      }
    }
    wave_lds_sync();  // guard sA/sB reuse next iteration
  }
}

extern "C" void kernel_launch(void* const* d_in, const int* in_sizes, int n_in,
                              void* d_out, int out_size, void* d_ws, size_t ws_size,
                              hipStream_t stream) {
  (void)in_sizes; (void)n_in; (void)out_size; (void)d_ws; (void)ws_size;
  const int*   x       = (const int*)  d_in[0];
  const float* tok_emb = (const float*)d_in[1];
  const float* pos_emb = (const float*)d_in[2];
  const float* Wq      = (const float*)d_in[3];
  const float* Wk      = (const float*)d_in[4];
  const float* Wv      = (const float*)d_in[5];
  const float* W1      = (const float*)d_in[6];
  const float* b1      = (const float*)d_in[7];
  const float* Wlm     = (const float*)d_in[8];
  const float* blm     = (const float*)d_in[9];
  float*       out     = (float*)d_out;

  fused_bigram_attn_kernel<<<dim3(NBLK), dim3(WPB * 32), 0, stream>>>(
      x, tok_emb, pos_emb, Wq, Wk, Wv, W1, b1, Wlm, blm, out);
}
